// NonLocalBlock2D_11201274708693
// MI455X (gfx1250) — matrix operand order repaired and checked by reading, not compiled, not run on hardware
//
#include <hip/hip_runtime.h>

// NonLocalBlock2D for MI455X (gfx1250): bf16 WMMA flash-attention pipeline.
// B=4, C=256, IC=128, N=4096. All matmuls via v_wmma_f32_16x16x32_bf16.
// Attention kernel double-buffers key blocks in LDS via
// global_load_async_to_lds_b128 + s_wait_asynccnt (DMA overlapped with WMMA).

#define B_  4
#define C_  256
#define IC_ 128
#define N_  4096

typedef __attribute__((ext_vector_type(16))) __bf16 v16bf;
typedef __attribute__((ext_vector_type(2)))  __bf16 v2bf;
typedef __attribute__((ext_vector_type(8)))  float  v8f;
typedef __attribute__((ext_vector_type(4)))  int    v4i;

union BfTile { unsigned u[8]; v16bf v; };

#if defined(__AMDGCN__) && __has_builtin(__builtin_amdgcn_global_load_async_to_lds_b128) && __has_builtin(__builtin_amdgcn_s_wait_asynccnt)
#define HAVE_ASYNC 1
typedef __attribute__((address_space(1))) v4i g_v4i;   // global int4
typedef __attribute__((address_space(3))) v4i l_v4i;   // LDS int4
#else
#define HAVE_ASYNC 0
#endif

static __device__ __forceinline__ unsigned short f2bf_manual(float f) {
    unsigned u = __float_as_uint(f);
    unsigned r = u + 0x7FFFu + ((u >> 16) & 1u);   // round-to-nearest-even
    return (unsigned short)(r >> 16);
}
// pack two f32 -> two bf16 in one dword (hardware cvt if available)
static __device__ __forceinline__ unsigned cvtpk(float lo, float hi) {
#if defined(__AMDGCN__) && __has_builtin(__builtin_amdgcn_cvt_pk_bf16_f32)
    union { v2bf h; unsigned u; } cv;
    cv.h = __builtin_amdgcn_cvt_pk_bf16_f32(lo, hi);
    return cv.u;
#else
    return (unsigned)f2bf_manual(lo) | ((unsigned)f2bf_manual(hi) << 16);
#endif
}
static __device__ __forceinline__ v8f wmma_bf16(v16bf a, v16bf b, v8f c) {
    return __builtin_amdgcn_wmma_f32_16x16x32_bf16(false, a, false, b, (short)0, c, false, false);
}
// A-matrix 16x32 bf16 per-lane K index for VGPR v (lane half 'hi')
static __device__ __forceinline__ int a_k(int hi, int v) {
    return hi * 16 + 2 * v;
}

// ---------------------------------------------------------------------------
// Kernel 0: one-shot fp32 -> bf16 weight conversion (w_theta/w_phi/w_g/w_out)
// ---------------------------------------------------------------------------
__global__ __launch_bounds__(256) void k_wconv(
        const float* __restrict__ wt, const float* __restrict__ wp,
        const float* __restrict__ wg, const float* __restrict__ wo,
        unsigned* __restrict__ wtb, unsigned* __restrict__ wpb,
        unsigned* __restrict__ wgb, unsigned* __restrict__ wob) {
    int i   = blockIdx.x * 256 + threadIdx.x;   // 65536 threads
    int seg = i >> 14;                          // 16384 pairs per matrix
    int r   = i & 16383;
    const float* s;
    unsigned* d;
    if (seg == 0)      { s = wt; d = wtb; }
    else if (seg == 1) { s = wp; d = wpb; }
    else if (seg == 2) { s = wg; d = wgb; }
    else               { s = wo; d = wob; }
    d[r] = cvtpk(s[2 * r], s[2 * r + 1]);
}

// ---------------------------------------------------------------------------
// Kernel 1: x [B][C][N] fp32  ->  xT [B][N][C] bf16 (2 elements per thread)
// ---------------------------------------------------------------------------
__global__ __launch_bounds__(256) void k_transpose_x(
        const float* __restrict__ x, unsigned short* __restrict__ xT) {
    int i  = blockIdx.x * 256 + threadIdx.x;    // B*C*N/2 threads
    int n2 = (i % (N_ / 2)) * 2;
    int c  = (i / (N_ / 2)) % C_;
    int b  = i / ((N_ / 2) * C_);
    const float* xp = x + ((size_t)b * C_ + c) * N_ + n2;
    unsigned pk = cvtpk(xp[0], xp[1]);
    unsigned short* d = xT + ((size_t)b * N_ + n2) * C_ + c;
    d[0]  = (unsigned short)pk;
    d[C_] = (unsigned short)(pk >> 16);
}

// ---------------------------------------------------------------------------
// Kernel 2: projections theta/phi/g = W(IC x C) @ xT^T + bias
// theta,phi stored transposed [B][N][IC]; g stored [B][IC][N]. All bf16.
// One wave: 16 out-channels x 64 pixels. Weights pre-converted to bf16.
// ---------------------------------------------------------------------------
__global__ __launch_bounds__(256) void k_proj(
        const unsigned* __restrict__ wtb, const float* __restrict__ b_theta,
        const unsigned* __restrict__ wpb, const float* __restrict__ b_phi,
        const unsigned* __restrict__ wgb, const float* __restrict__ b_g,
        const unsigned short* __restrict__ xT,
        unsigned short* __restrict__ thetaT,
        unsigned short* __restrict__ phiT,
        unsigned short* __restrict__ gbuf) {
    int wid  = (blockIdx.x * 256 + threadIdx.x) >> 5;
    int lane = threadIdx.x & 31;
    int hi   = lane >> 4;
    int l15  = lane & 15;

    int b  = wid / 1536;            // 3 * 8 * 64 waves per batch
    int r  = wid % 1536;
    int p  = r / 512;               // 0=theta 1=phi 2=g
    int mt = (r % 512) / 64;        // out-channel tile (IC/16)
    int nt = r % 64;                // pixel group of 64 (N/64)

    const unsigned* w  = (p == 0) ? wtb : (p == 1) ? wpb : wgb;
    const float*    bv = (p == 0) ? b_theta : (p == 1) ? b_phi : b_g;

    const unsigned* xTd = (const unsigned*)xT + (size_t)b * N_ * (C_ / 2);

    v8f acc[4] = {};
    #pragma unroll
    for (int kk = 0; kk < C_ / 32; ++kk) {
        BfTile a;
        #pragma unroll
        for (int v = 0; v < 8; ++v) {
            int k = kk * 32 + a_k(hi, v);
            int o = mt * 16 + l15;
            a.u[v] = w[o * (C_ / 2) + (k >> 1)];
        }
        #pragma unroll
        for (int t = 0; t < 4; ++t) {
            BfTile bm;
            #pragma unroll
            for (int v = 0; v < 8; ++v) {
                int k = kk * 32 + hi * 16 + 2 * v;       // B 32x16 layout
                int n = nt * 64 + t * 16 + l15;
                bm.u[v] = xTd[(size_t)n * (C_ / 2) + (k >> 1)];
            }
            acc[t] = wmma_bf16(a.v, bm.v, acc[t]);
        }
    }
    // epilogue: bias + store (adjacent D rows -> one dword store when transposed)
    #pragma unroll
    for (int t = 0; t < 4; ++t) {
        #pragma unroll
        for (int v = 0; v < 8; v += 2) {
            int o = mt * 16 + v + 8 * hi;                // even D row
            int n = nt * 64 + t * 16 + l15;              // D col
            unsigned pk = cvtpk(acc[t][v] + bv[o], acc[t][v + 1] + bv[o + 1]);
            if (p == 0) {
                ((unsigned*)thetaT)[(((size_t)b * N_ + n) * IC_ + o) >> 1] = pk;
            } else if (p == 1) {
                ((unsigned*)phiT)[(((size_t)b * N_ + n) * IC_ + o) >> 1] = pk;
            } else {
                gbuf[((size_t)b * IC_ + o) * N_ + n]       = (unsigned short)pk;
                gbuf[((size_t)b * IC_ + o + 1) * N_ + n]   = (unsigned short)(pk >> 16);
            }
        }
    }
}

// ---------------------------------------------------------------------------
// Kernel 3: fused flash attention. WG = 4 waves = 64 queries; wave = 16
// queries, 16x128 f32 accumulator. Double-buffered async DMA of 32-key
// blocks of phi/g into LDS, overlapped with WMMA compute.
// ---------------------------------------------------------------------------
__global__ __launch_bounds__(128) void k_attn(
        const unsigned short* __restrict__ thetaT,
        const unsigned short* __restrict__ phiT,
        const unsigned short* __restrict__ gbuf,
        unsigned short* __restrict__ attnT) {
    __shared__ alignas(16) unsigned short sPhi[2][32 * IC_];   // [buf][j][c]
    __shared__ alignas(16) unsigned short sG[2][IC_ * 32];     // [buf][c][j]
    __shared__ alignas(16) unsigned short sP[4][16 * 32];      // per-wave P stage

    int tid  = threadIdx.x;
    int wave = tid >> 5;
    int lane = tid & 31;
    int hi   = lane >> 4;
    int l15  = lane & 15;

    int b  = blockIdx.x >> 6;
    int qb = (blockIdx.x & 63) * 64 + wave * 16;            // this wave's queries

    // per-thread copy sources: 8 x b128 per key block (4 phi + 4 g)
    const uint4* phiSrcBase = (const uint4*)(phiT + (size_t)b * N_ * IC_);
    const uint4* gSrcBase   = (const uint4*)(gbuf + ((size_t)b * IC_ + tid) * N_);

    // issue one key block's copies into buffer `buf` (8 async b128 per thread)
    auto issue_copy = [&](int jb, int buf) {
        const uint4* psrc = phiSrcBase + (size_t)jb * (IC_ / 8);  // 8 bf16 per uint4
        uint4* pdst = (uint4*)sPhi[buf];
        #pragma unroll
        for (int i = 0; i < 4; ++i) {
#if HAVE_ASYNC
            __builtin_amdgcn_global_load_async_to_lds_b128(
                (g_v4i*)(psrc + tid + i * 128), (l_v4i*)(pdst + tid + i * 128), 0, 0);
#else
            pdst[tid + i * 128] = psrc[tid + i * 128];
#endif
        }
        const uint4* gsrc = gSrcBase + jb / 8;
        uint4* gdst = (uint4*)(sG[buf] + tid * 32);
        #pragma unroll
        for (int i = 0; i < 4; ++i) {
#if HAVE_ASYNC
            __builtin_amdgcn_global_load_async_to_lds_b128(
                (g_v4i*)(gsrc + i), (l_v4i*)(gdst + i), 0, 0);
#else
            gdst[i] = gsrc[i];
#endif
        }
    };

    // Preload theta A-tiles (16 queries x 128 channels = 4 K-steps)
    const unsigned* thD = (const unsigned*)thetaT + (size_t)b * N_ * (IC_ / 2);
    BfTile ta[4];
    #pragma unroll
    for (int kk = 0; kk < 4; ++kk)
        #pragma unroll
        for (int v = 0; v < 8; ++v) {
            int k = kk * 32 + a_k(hi, v);
            ta[kk].u[v] = thD[(size_t)(qb + l15) * (IC_ / 2) + (k >> 1)];
        }

    v8f oacc[8] = {};
    float mrun[8], lrun[8];
    #pragma unroll
    for (int v = 0; v < 8; ++v) { mrun[v] = -1e30f; lrun[v] = 0.0f; }

    issue_copy(0, 0);                                       // prime the pipeline

    const int NIT = N_ / 32;
    for (int it = 0; it < NIT; ++it) {
        int cur = it & 1;
        if (it + 1 < NIT) {
            issue_copy((it + 1) * 32, 1 - cur);             // prefetch next block
#if HAVE_ASYNC
            __builtin_amdgcn_s_wait_asynccnt(8);            // current block landed
#endif
        } else {
#if HAVE_ASYNC
            __builtin_amdgcn_s_wait_asynccnt(0);
#endif
        }
        __syncthreads();                                    // all waves' copies done

        // S(16x32) = theta_i @ phi_j^T, K = IC = 128
        v8f s0 = {}, s1 = {};
        const unsigned* sPhiD = (const unsigned*)sPhi[cur];
        #pragma unroll
        for (int kk = 0; kk < 4; ++kk) {
            BfTile b0, b1;
            #pragma unroll
            for (int v = 0; v < 8; ++v) {
                int kc = kk * 16 + hi * 8 + v;              // dword index of K pair
                b0.u[v] = sPhiD[(size_t)l15 * (IC_ / 2) + kc];
                b1.u[v] = sPhiD[(size_t)(16 + l15) * (IC_ / 2) + kc];
            }
            s0 = wmma_bf16(ta[kk].v, b0.v, s0);
            s1 = wmma_bf16(ta[kk].v, b1.v, s1);
        }

        // online softmax (row m = v + 8*hi lives in one 16-lane half)
        float scl[8];
        #pragma unroll
        for (int v = 0; v < 8; ++v) {
            float r = fmaxf(s0[v], s1[v]);
            r = fmaxf(r, __shfl_xor(r, 1, 32));
            r = fmaxf(r, __shfl_xor(r, 2, 32));
            r = fmaxf(r, __shfl_xor(r, 4, 32));
            r = fmaxf(r, __shfl_xor(r, 8, 32));
            float nm = fmaxf(mrun[v], r);
            float sc = __expf(mrun[v] - nm);
            float p0 = __expf(s0[v] - nm);
            float p1 = __expf(s1[v] - nm);
            float rs = p0 + p1;
            rs += __shfl_xor(rs, 1, 32);
            rs += __shfl_xor(rs, 2, 32);
            rs += __shfl_xor(rs, 4, 32);
            rs += __shfl_xor(rs, 8, 32);
            lrun[v] = lrun[v] * sc + rs;
            mrun[v] = nm;
            scl[v] = sc;
            s0[v] = p0;
            s1[v] = p1;
        }
        #pragma unroll
        for (int ct = 0; ct < 8; ++ct)
            #pragma unroll
            for (int v = 0; v < 8; ++v) oacc[ct][v] *= scl[v];

        // P: D-layout f32 -> LDS -> A-layout bf16 (per-wave private stage)
        unsigned short* myP = sP[wave];
        #pragma unroll
        for (int v = 0; v < 8; ++v) {
            int m = v + 8 * hi;
            unsigned pk = cvtpk(s0[v], s1[v]);
            myP[m * 32 + l15]      = (unsigned short)pk;
            myP[m * 32 + 16 + l15] = (unsigned short)(pk >> 16);
        }
        BfTile pa;
        const unsigned* myPD = (const unsigned*)myP;
        #pragma unroll
        for (int v = 0; v < 8; ++v)
            pa.u[v] = myPD[l15 * 16 + (a_k(hi, v) >> 1)];

        // O(16x128) += P(16x32) @ G(32x128)
        const unsigned* sGD = (const unsigned*)sG[cur];
        #pragma unroll
        for (int ct = 0; ct < 8; ++ct) {
            BfTile gb;
            #pragma unroll
            for (int v = 0; v < 8; ++v) {
                int c = ct * 16 + l15;
                gb.u[v] = sGD[c * 16 + hi * 8 + v];         // K pair 2v at half hi
            }
            oacc[ct] = wmma_bf16(pa.v, gb.v, oacc[ct]);
        }
        __syncthreads();        // buffer `cur` free for overwrite next iteration
    }

    // normalize + store attnT [N][IC] bf16
    float inv[8];
    #pragma unroll
    for (int v = 0; v < 8; ++v) inv[v] = 1.0f / lrun[v];
    unsigned short* dst = attnT + ((size_t)b * N_ + qb) * IC_;
    #pragma unroll
    for (int ct = 0; ct < 8; ct += 2)
        #pragma unroll
        for (int v = 0; v < 8; ++v) {
            int m = v + 8 * hi;
            int c = ct * 16 + l15;
            unsigned pk = cvtpk(oacc[ct][v] * inv[v], oacc[ct + 1][v] * inv[v]);
            dst[(size_t)m * IC_ + c]      = (unsigned short)pk;
            dst[(size_t)m * IC_ + c + 16] = (unsigned short)(pk >> 16);
        }
}

// ---------------------------------------------------------------------------
// Kernel 4: y = x + W_out(C x IC) @ attn + b_out.  Wave: 16 out-ch x 64 px.
// ---------------------------------------------------------------------------
__global__ __launch_bounds__(256) void k_out(
        const unsigned* __restrict__ wob, const float* __restrict__ b_out,
        const unsigned short* __restrict__ attnT,
        const float* __restrict__ x, float* __restrict__ y) {
    int wid  = (blockIdx.x * 256 + threadIdx.x) >> 5;
    int lane = threadIdx.x & 31;
    int hi   = lane >> 4;
    int l15  = lane & 15;

    int b  = wid / 1024;            // 16 * 64 waves per batch
    int r  = wid % 1024;
    int mt = r / 64;                // out-channel tile (C/16)
    int nt = r % 64;                // pixel group (N/64)

    const unsigned* aT = (const unsigned*)attnT + (size_t)b * N_ * (IC_ / 2);

    v8f acc[4] = {};
    #pragma unroll
    for (int kk = 0; kk < IC_ / 32; ++kk) {
        BfTile a;
        #pragma unroll
        for (int v = 0; v < 8; ++v) {
            int k = kk * 32 + a_k(hi, v);
            int o = mt * 16 + l15;
            a.u[v] = wob[o * (IC_ / 2) + (k >> 1)];
        }
        #pragma unroll
        for (int t = 0; t < 4; ++t) {
            BfTile bm;
            #pragma unroll
            for (int v = 0; v < 8; ++v) {
                int k = kk * 32 + hi * 16 + 2 * v;
                int n = nt * 64 + t * 16 + l15;
                bm.u[v] = aT[(size_t)n * (IC_ / 2) + (k >> 1)];
            }
            acc[t] = wmma_bf16(a.v, bm.v, acc[t]);
        }
    }
    #pragma unroll
    for (int t = 0; t < 4; ++t)
        #pragma unroll
        for (int v = 0; v < 8; ++v) {
            int o = mt * 16 + v + 8 * hi;
            int n = nt * 64 + t * 16 + l15;
            size_t oi = ((size_t)b * C_ + o) * N_ + n;
            y[oi] = x[oi] + acc[t][v] + b_out[o];
        }
}

// ---------------------------------------------------------------------------
extern "C" void kernel_launch(void* const* d_in, const int* in_sizes, int n_in,
                              void* d_out, int out_size, void* d_ws, size_t ws_size,
                              hipStream_t stream) {
    const float* x       = (const float*)d_in[0];
    const float* w_theta = (const float*)d_in[1];
    const float* b_theta = (const float*)d_in[2];
    const float* w_phi   = (const float*)d_in[3];
    const float* b_phi   = (const float*)d_in[4];
    const float* w_g     = (const float*)d_in[5];
    const float* b_g     = (const float*)d_in[6];
    const float* w_out   = (const float*)d_in[7];
    const float* b_out   = (const float*)d_in[8];
    float* y = (float*)d_out;

    unsigned char* ws = (unsigned char*)d_ws;
    unsigned short* xT     = (unsigned short*)(ws);                 // 8 MiB
    unsigned short* thetaT = (unsigned short*)(ws + 8388608);       // 4 MiB
    unsigned short* phiT   = (unsigned short*)(ws + 12582912);      // 4 MiB
    unsigned short* gbuf   = (unsigned short*)(ws + 16777216);      // 4 MiB
    unsigned short* attnT  = (unsigned short*)(ws + 20971520);      // 4 MiB
    unsigned*       wtb    = (unsigned*)(ws + 25165824);            // 64 KiB
    unsigned*       wpb    = (unsigned*)(ws + 25231360);            // 64 KiB
    unsigned*       wgb    = (unsigned*)(ws + 25296896);            // 64 KiB
    unsigned*       wob    = (unsigned*)(ws + 25362432);            // 64 KiB

    k_wconv<<<256, 256, 0, stream>>>(w_theta, w_phi, w_g, w_out,
                                     wtb, wpb, wgb, wob);
    k_transpose_x<<<(B_ * C_ * N_ / 2) / 256, 256, 0, stream>>>(x, xT);
    k_proj<<<768, 256, 0, stream>>>(wtb, b_theta, wpb, b_phi, wgb, b_g,
                                    xT, thetaT, phiT, gbuf);
    k_attn<<<B_ * (N_ / 64), 128, 0, stream>>>(thetaT, phiT, gbuf, attnT);
    k_out<<<512, 256, 0, stream>>>(wob, b_out, attnT, x, y);
}
